// MambaDynamics_7945689498198
// MI455X (gfx1250) — matrix operand order repaired and checked by reading, not compiled, not run on hardware
//
#include <hip/hip_runtime.h>
#include <hip/hip_bf16.h>
#include <math.h>

// ---------------------------------------------------------------------------
// MambaDynamics for MI455X (gfx1250, wave32, WMMA).
// GEMMs: f16 WMMA (v_wmma_f32_16x16x32_f16), f32 accumulate, compile-time
// shapes so no EXEC-divergent guards surround the WMMAs.
// ---------------------------------------------------------------------------

#define BB   4
#define TT   2048
#define BT   (BB*TT)          // 8192 rows
#define LAT  256
#define OBJ  64
#define DMOD 256
#define NL   4
#define DSN  16               // state dim
#define DCV  4                // conv taps
#define DI   512
#define DTR  16
#define KIN  (LAT+OBJ)        // 320
#define NPROJ (DTR + 2*DSN)   // 48

typedef __attribute__((ext_vector_type(16))) _Float16 v16h;
typedef __attribute__((ext_vector_type(8)))  _Float16 v8h;
typedef __attribute__((ext_vector_type(8)))  float    v8f;

union Frag16 { v16h v; v8h h[2]; };

// ---------------------------------------------------------------------------
// Elementwise helpers
// ---------------------------------------------------------------------------
__global__ void cvt_f32_to_f16_kernel(const float* __restrict__ src,
                                      _Float16* __restrict__ dst, int n) {
    int i = blockIdx.x * blockDim.x + threadIdx.x;
    if (i < n) dst[i] = (_Float16)src[i];
}

// Build packed input A = [z_seq | o_seq] as f16, shape [BT x 320]
__global__ void pack_input_kernel(const float* __restrict__ z,
                                  const float* __restrict__ o,
                                  _Float16* __restrict__ dst) {
    int i = blockIdx.x * blockDim.x + threadIdx.x;
    if (i >= BT * KIN) return;
    int r = i / KIN, c = i % KIN;
    float v = (c < LAT) ? z[(size_t)r * LAT + c] : o[(size_t)r * OBJ + (c - LAT)];
    dst[i] = (_Float16)v;
}

// ---------------------------------------------------------------------------
// WMMA GEMM:  C[BT x N] (f32) = A[BT x K](f16) * W[N x K](f16)^T  (+bias)
// EPI==0: C = acc (+bias);  EPI==1: C += acc (residual accumulate).
//
// Compile-time shapes: no guards, K-loop fully unrolled, WMMAs never sit
// under exec masking. Each wave computes a (MSUB*16) x (NSUB*16) patch:
// A fragments are reused across all NSUB B fragments.
// Launch: block = WAVES*32 threads; grid = (BT/(16*MSUB), N/(WAVES*NSUB*16)).
// ---------------------------------------------------------------------------
template <int EPI, int N, int K, int NSUB, int MSUB>
__global__ void gemm_f16_wmma_kernel(const _Float16* __restrict__ A,
                                     const _Float16* __restrict__ W,
                                     const float* __restrict__ bias,
                                     float* __restrict__ C) {
    const int lane  = threadIdx.x & 31;
    const int wave  = __builtin_amdgcn_readfirstlane(threadIdx.x >> 5);
    const int waves = blockDim.x >> 5;
    const int mBase = blockIdx.x * (MSUB * 16);
    const int nBase = (blockIdx.y * waves + wave) * (NSUB * 16);

    // Fragment layouts per ISA 7.12.2 (16-bit, wave32):
    //  A 16x32: row = lane&15; halves 0-7 -> K = 8*(lane>>4)+0..7,
    //           halves 8-15 -> K = 8*(lane>>4)+16..23.
    //  B 32x16: col = lane&15; same K mapping as A.
    //  C/D:     VGPR r -> row 8*(lane>>4)+r, col = lane&15.
    const int aKoff = (lane >> 4) * 8;
    const int bKoff = (lane >> 4) * 8;

    size_t aRow[MSUB];
#pragma unroll
    for (int i = 0; i < MSUB; ++i)
        aRow[i] = (size_t)(mBase + i * 16 + (lane & 15)) * K;
    const size_t wRow0 = (size_t)(nBase + (lane & 15)) * K;

    v8f acc[MSUB][NSUB];
#pragma unroll
    for (int i = 0; i < MSUB; ++i)
#pragma unroll
        for (int j = 0; j < NSUB; ++j) acc[i][j] = (v8f){};

#pragma unroll
    for (int k = 0; k < K; k += 32) {
        Frag16 af[MSUB];
#pragma unroll
        for (int i = 0; i < MSUB; ++i) {
            af[i].h[0] = *(const v8h*)(A + aRow[i] + k + aKoff);
            af[i].h[1] = *(const v8h*)(A + aRow[i] + k + aKoff + 16);
        }
#pragma unroll
        for (int j = 0; j < NSUB; ++j) {
            Frag16 bf;
            const size_t wRow = wRow0 + (size_t)j * 16 * K;
            bf.h[0] = *(const v8h*)(W + wRow + k + bKoff);
            bf.h[1] = *(const v8h*)(W + wRow + k + bKoff + 16);
#pragma unroll
            for (int i = 0; i < MSUB; ++i) {
                acc[i][j] = __builtin_amdgcn_wmma_f32_16x16x32_f16(
                    /*neg_a=*/false, af[i].v, /*neg_b=*/false, bf.v,
                    /*c_mod=*/(short)0, acc[i][j],
                    /*reuse_a=*/false, /*reuse_b=*/false);
            }
        }
    }

#pragma unroll
    for (int j = 0; j < NSUB; ++j) {
        const int nc = nBase + j * 16 + (lane & 15);
        const float bv = (EPI == 0 && bias) ? bias[nc] : 0.0f;
#pragma unroll
        for (int i = 0; i < MSUB; ++i) {
#pragma unroll
            for (int r = 0; r < 8; ++r) {
                const int mr = mBase + i * 16 + (lane >> 4) * 8 + r;
                const size_t idx = (size_t)mr * N + nc;
                const float v = acc[i][j][r] + bv;
                if (EPI == 1) C[idx] = C[idx] + v;
                else          C[idx] = v;
            }
        }
    }
}

// ---------------------------------------------------------------------------
// Depthwise causal conv (DC=4) + SiLU over x-half of xz. Writes f16.
// ---------------------------------------------------------------------------
__global__ void conv_silu_kernel(const float* __restrict__ xz,
                                 const float* __restrict__ cw,  // [DI x DC]
                                 const float* __restrict__ cb,  // [DI]
                                 _Float16* __restrict__ out) {
    int i = blockIdx.x * blockDim.x + threadIdx.x;
    if (i >= BT * DI) return;
    int r = i / DI, d = i % DI;
    int t = r % TT;
    float s = cb[d];
#pragma unroll
    for (int k = 0; k < DCV; ++k) {
        int dt = t - (DCV - 1) + k;               // left pad DC-1
        if (dt >= 0) s += xz[(size_t)(r - (DCV - 1) + k) * (2 * DI) + d] * cw[d * DCV + k];
    }
    float sig = 1.0f / (1.0f + __expf(-s));
    out[i] = (_Float16)(s * sig);
}

// ---------------------------------------------------------------------------
// Selective scan, fused: dt = softplus(dt_r @ dt_w^T + dt_b); recurrence over
// T; y = (scan + x*D) * silu(z); writes f16 for the out_w GEMM.
// ---------------------------------------------------------------------------
__global__ void scan_kernel(const _Float16* __restrict__ xc,   // [BT x DI]
                            const float* __restrict__ proj,    // [BT x 48]
                            const float* __restrict__ xz,      // [BT x 2*DI]
                            const float* __restrict__ dt_w,    // [DI x DTR]
                            const float* __restrict__ dt_b,    // [DI]
                            const float* __restrict__ A_log,   // [DI x DSN]
                            const float* __restrict__ Dv,      // [DI]
                            _Float16* __restrict__ yg) {       // [BT x DI]
    const int d = blockIdx.x * blockDim.x + threadIdx.x;
    const int b = blockIdx.y;

    float Arow[DSN], dtw[DTR], st[DSN];
#pragma unroll
    for (int n = 0; n < DSN; ++n) {
        Arow[n] = -__expf(A_log[(size_t)d * DSN + n]);
        st[n] = 0.0f;
    }
#pragma unroll
    for (int j = 0; j < DTR; ++j) dtw[j] = dt_w[(size_t)d * DTR + j];
    const float Dd = Dv[d];
    const float dtb = dt_b[d];

    __shared__ float sP[NPROJ];   // [dt_r | B | C]

    for (int t = 0; t < TT; ++t) {
        const size_t r = (size_t)b * TT + t;
        __syncthreads();
        if (threadIdx.x < NPROJ) sP[threadIdx.x] = proj[r * NPROJ + threadIdx.x];
        __syncthreads();

        const float xv = (float)xc[r * DI + d];
        const float zv = xz[r * (2 * DI) + DI + d];

        float s = dtb;
#pragma unroll
        for (int j = 0; j < DTR; ++j) s += sP[j] * dtw[j];
        const float dt = (s > 20.0f) ? s : __logf(1.0f + __expf(s));

        float acc = xv * Dd;
        const float dtx = dt * xv;
#pragma unroll
        for (int n = 0; n < DSN; ++n) {
            float dA = __expf(dt * Arow[n]);
            st[n] = dA * st[n] + dtx * sP[DTR + n];
            acc += st[n] * sP[DTR + DSN + n];
        }
        const float sig = 1.0f / (1.0f + __expf(-zv));
        yg[r * DI + d] = (_Float16)(acc * (zv * sig));
    }
}

// ---------------------------------------------------------------------------
// LayerNorm over DM=256 per row; writes f16 for head GEMMs.
// ---------------------------------------------------------------------------
__global__ void layernorm_kernel(const float* __restrict__ h,
                                 const float* __restrict__ g,
                                 const float* __restrict__ bta,
                                 _Float16* __restrict__ out) {
    const int r = blockIdx.x;
    const int c = threadIdx.x;          // 256
    __shared__ float red[DMOD];
    const float v = h[(size_t)r * DMOD + c];
    red[c] = v; __syncthreads();
    for (int s = DMOD / 2; s > 0; s >>= 1) {
        if (c < s) red[c] += red[c + s];
        __syncthreads();
    }
    const float mu = red[0] * (1.0f / DMOD);
    __syncthreads();
    const float dv = v - mu;
    red[c] = dv * dv; __syncthreads();
    for (int s = DMOD / 2; s > 0; s >>= 1) {
        if (c < s) red[c] += red[c + s];
        __syncthreads();
    }
    const float var = red[0] * (1.0f / DMOD);
    out[(size_t)r * DMOD + c] = (_Float16)(dv * rsqrtf(var + 1e-5f) * g[c] + bta[c]);
}

// ---------------------------------------------------------------------------
// Host side
// ---------------------------------------------------------------------------
static inline size_t alignUp(size_t x) { return (x + 255) & ~(size_t)255; }

extern "C" void kernel_launch(void* const* d_in, const int* in_sizes, int n_in,
                              void* d_out, int out_size, void* d_ws, size_t ws_size,
                              hipStream_t stream) {
    const float* z_seq     = (const float*)d_in[0];
    const float* o_seq     = (const float*)d_in[1];
    const float* in_proj_w = (const float*)d_in[2];
    const float* in_proj_b = (const float*)d_in[3];
    const float* blk_in_w  = (const float*)d_in[4];
    const float* blk_cw    = (const float*)d_in[5];
    const float* blk_cb    = (const float*)d_in[6];
    const float* blk_xp_w  = (const float*)d_in[7];
    const float* blk_dt_w  = (const float*)d_in[8];
    const float* blk_dt_b  = (const float*)d_in[9];
    const float* blk_A_log = (const float*)d_in[10];
    const float* blk_D     = (const float*)d_in[11];
    const float* blk_out_w = (const float*)d_in[12];
    const float* ln_g      = (const float*)d_in[13];
    const float* ln_b      = (const float*)d_in[14];
    const float* out_lat_w = (const float*)d_in[15];
    const float* out_lat_b = (const float*)d_in[16];
    const float* out_obj_w = (const float*)d_in[17];
    const float* out_obj_b = (const float*)d_in[18];
    float* out = (float*)d_out;

    // ---- workspace carve-up (all offsets 256B aligned) ----
    char* ws = (char*)d_ws;
    size_t off = 0;
    _Float16* Ain     = (_Float16*)(ws + off); off += alignUp((size_t)BT * KIN * 2);
    _Float16* w_in    = (_Float16*)(ws + off); off += alignUp((size_t)DMOD * KIN * 2);
    _Float16* w_blkin = (_Float16*)(ws + off); off += alignUp((size_t)NL * 2 * DI * DMOD * 2);
    _Float16* w_xp    = (_Float16*)(ws + off); off += alignUp((size_t)NL * NPROJ * DI * 2);
    _Float16* w_outw  = (_Float16*)(ws + off); off += alignUp((size_t)NL * DMOD * DI * 2);
    _Float16* w_lat   = (_Float16*)(ws + off); off += alignUp((size_t)LAT * DMOD * 2);
    _Float16* w_obj   = (_Float16*)(ws + off); off += alignUp((size_t)OBJ * DMOD * 2);
    float*    h       = (float*)   (ws + off); off += alignUp((size_t)BT * DMOD * 4);
    _Float16* h16     = (_Float16*)(ws + off); off += alignUp((size_t)BT * DMOD * 2);
    float*    xz      = (float*)   (ws + off); off += alignUp((size_t)BT * 2 * DI * 4);
    _Float16* xc16    = (_Float16*)(ws + off); off += alignUp((size_t)BT * DI * 2);
    float*    proj    = (float*)   (ws + off); off += alignUp((size_t)BT * NPROJ * 4);
    _Float16* yg16    = (_Float16*)(ws + off); off += alignUp((size_t)BT * DI * 2);
    _Float16* hn16    = (_Float16*)(ws + off); off += alignUp((size_t)BT * DMOD * 2);
    (void)off; (void)ws_size;

    const int TB = 256;
    auto g1 = [&](size_t n) { return dim3((unsigned)((n + TB - 1) / TB)); };

    // ---- weight conversions (f32 -> f16) ----
    { size_t n = (size_t)DMOD * KIN;
      cvt_f32_to_f16_kernel<<<g1(n), TB, 0, stream>>>(in_proj_w, w_in, (int)n); }
    { size_t n = (size_t)NL * 2 * DI * DMOD;
      cvt_f32_to_f16_kernel<<<g1(n), TB, 0, stream>>>(blk_in_w, w_blkin, (int)n); }
    { size_t n = (size_t)NL * NPROJ * DI;
      cvt_f32_to_f16_kernel<<<g1(n), TB, 0, stream>>>(blk_xp_w, w_xp, (int)n); }
    { size_t n = (size_t)NL * DMOD * DI;
      cvt_f32_to_f16_kernel<<<g1(n), TB, 0, stream>>>(blk_out_w, w_outw, (int)n); }
    { size_t n = (size_t)LAT * DMOD;
      cvt_f32_to_f16_kernel<<<g1(n), TB, 0, stream>>>(out_lat_w, w_lat, (int)n); }
    { size_t n = (size_t)OBJ * DMOD;
      cvt_f32_to_f16_kernel<<<g1(n), TB, 0, stream>>>(out_obj_w, w_obj, (int)n); }

    // ---- input pack + input projection ----
    pack_input_kernel<<<g1((size_t)BT * KIN), TB, 0, stream>>>(z_seq, o_seq, Ain);

    // Launch geometry helper: grid = (BT/(16*MSUB), N/(waves*NSUB*16))
    // in_proj: N=256,K=320, 4 waves x NSUB4 x MSUB2 -> grid (256,1)
    gemm_f16_wmma_kernel<0, DMOD, KIN, 4, 2><<<dim3(BT/32, 1), 128, 0, stream>>>(
        Ain, w_in, in_proj_b, h);

    // ---- Mamba layers ----
    for (int l = 0; l < NL; ++l) {
        cvt_f32_to_f16_kernel<<<g1((size_t)BT * DMOD), TB, 0, stream>>>(
            h, h16, BT * DMOD);

        // xz = h @ in_w^T : N=1024, K=256 -> grid (256, 4)
        gemm_f16_wmma_kernel<0, 2*DI, DMOD, 4, 2><<<dim3(BT/32, 4), 128, 0, stream>>>(
            h16, w_blkin + (size_t)l * 2 * DI * DMOD, nullptr, xz);

        conv_silu_kernel<<<g1((size_t)BT * DI), TB, 0, stream>>>(
            xz, blk_cw + (size_t)l * DI * DCV, blk_cb + (size_t)l * DI, xc16);

        // proj = x @ xp_w^T : N=48, K=512 -> 1 wave/block, NSUB=3 -> grid (256,1)
        gemm_f16_wmma_kernel<0, NPROJ, DI, 3, 2><<<dim3(BT/32, 1), 32, 0, stream>>>(
            xc16, w_xp + (size_t)l * NPROJ * DI, nullptr, proj);

        scan_kernel<<<dim3(DI / 256, BB), 256, 0, stream>>>(
            xc16, proj, xz,
            blk_dt_w + (size_t)l * DI * DTR, blk_dt_b + (size_t)l * DI,
            blk_A_log + (size_t)l * DI * DSN, blk_D + (size_t)l * DI, yg16);

        // h += yg @ out_w^T : N=256, K=512 -> grid (256,1)
        gemm_f16_wmma_kernel<1, DMOD, DI, 4, 2><<<dim3(BT/32, 1), 128, 0, stream>>>(
            yg16, w_outw + (size_t)l * DMOD * DI, nullptr, h);
    }

    // ---- LayerNorm + output heads ----
    layernorm_kernel<<<dim3(BT), DMOD, 0, stream>>>(h, ln_g, ln_b, hn16);

    // z_next: N=256, K=256 -> grid (256,1)
    gemm_f16_wmma_kernel<0, LAT, DMOD, 4, 2><<<dim3(BT/32, 1), 128, 0, stream>>>(
        hn16, w_lat, out_lat_b, out);
    // o_next: N=64, K=256 -> 1 wave/block, NSUB=4 -> grid (256,1)
    gemm_f16_wmma_kernel<0, OBJ, DMOD, 4, 2><<<dim3(BT/32, 1), 32, 0, stream>>>(
        hn16, w_obj, out_obj_b, out + (size_t)BT * LAT);
}
